// SwinTransformerBlock3D_19542101197166
// MI455X (gfx1250) — matrix operand order repaired and checked by reading, not compiled, not run on hardware
//
#include <hip/hip_runtime.h>
#include <hip/hip_bf16.h>
#include <math.h>

typedef _Float16 h16;
typedef __attribute__((ext_vector_type(16))) _Float16 v16h;
typedef __attribute__((ext_vector_type(8)))  _Float16 v8h;
typedef __attribute__((ext_vector_type(8)))  float    v8f;

#define DEVI __device__ __forceinline__

// ---------------- problem constants (fixed by the reference) ----------------
constexpr int Bz = 2, Dz = 8, Hz = 56, Wz = 56, Cz = 128;
constexpr int HEADS = 4, HD = 32;
constexpr int Nw = 392;              // 8*7*7 tokens per window
constexpr int Np = 416;              // padded to multiple of 32 (WMMA K)
constexpr int NWIN = Bz * 1 * 8 * 8; // 128 windows
constexpr int Mwin = NWIN * Np;      // 53248 padded window-token rows
constexpr int TOK = Bz * Dz * Hz * Wz; // 50176 tokens
constexpr int HID = 512;
constexpr float EPSL = 1e-3f;
constexpr float SCALE = 0.17677669529663687f; // 32^-0.5

DEVI v8f wmma32(v16h a, v16h b, v8f c) {
  return __builtin_amdgcn_wmma_f32_16x16x32_f16(false, a, false, b, (short)0, c,
                                                false, false);
}

// A fragment: row-major MxK f16. k order used here (contiguous, same as load_bT):
// lanes 0-15: K = k0+0..15; lanes 16-31: K = k0+16..31
DEVI v16h load_a(const h16* __restrict__ A, int lda, int m0, int k0) {
  int lane = threadIdx.x & 31;
  const h16* p = A + (size_t)(m0 + (lane & 15)) * lda + k0 + ((lane < 16) ? 0 : 16);
  v8h lo = *(const v8h*)p;
  v8h hi = *(const v8h*)(p + 8);
  return __builtin_shufflevector(lo, hi, 0, 1, 2, 3, 4, 5, 6, 7, 8, 9, 10, 11,
                                 12, 13, 14, 15);
}

// B fragment from an N-major (transposed) matrix: element B[kk][n] = T[n*ldk+kk].
// lanes 0-15 hold K=k0+0..15, lanes 16-31 hold K=k0+16..31 -> one contiguous
// 32-byte read per lane.
DEVI v16h load_bT(const h16* __restrict__ T, int ldk, int k0, int n0) {
  int lane = threadIdx.x & 31;
  return *(const v16h*)(T + (size_t)(n0 + (lane & 15)) * ldk + k0 +
                        ((lane < 16) ? 0 : 16));
}

DEVI int reg56(int p) { return p < 49 ? 0 : (p < 53 ? 1 : 2); } // H/W slice regions

// ---------------- weight convert + transpose: W(KxN f32) -> Wt(NxK f16) -----
__global__ void k_cvt_t(const float* __restrict__ s, h16* __restrict__ d, int K,
                        int N) {
  int i = blockIdx.x * blockDim.x + threadIdx.x;
  if (i >= K * N) return;
  int k = i / N, n = i % N;
  d[(size_t)n * K + k] = (h16)s[i];
}

// ---------------- LN1 + roll + window partition ----------------
__global__ void k_ln1_window(const float* __restrict__ x, const float* __restrict__ g,
                             const float* __restrict__ bt, h16* __restrict__ xw) {
  int wave = threadIdx.x >> 5, lane = threadIdx.x & 31;
  int t = blockIdx.x * (blockDim.x >> 5) + wave;
  if (t >= Mwin) return;
  int bn = t / Np, n = t % Np;
  h16* dst = xw + (size_t)t * Cz;
  if (n >= Nw) {
#pragma unroll
    for (int i = 0; i < 4; ++i) dst[lane * 4 + i] = (h16)0.f;
    return;
  }
  int b = bn >> 6, ih = (bn >> 3) & 7, iw = bn & 7;
  int td = n / 49, th = (n / 7) % 7, tw = n % 7;
  int d = td, hh = (ih * 7 + th + 3) % Hz, ww = (iw * 7 + tw + 3) % Wz;
  const float* src = x + ((((size_t)b * Dz + d) * Hz + hh) * Wz + ww) * Cz;
  float vals[4], s = 0.f, s2 = 0.f;
#pragma unroll
  for (int i = 0; i < 4; ++i) {
    float v = src[lane * 4 + i]; vals[i] = v; s += v; s2 += v * v;
  }
  for (int m = 16; m >= 1; m >>= 1) { s += __shfl_xor(s, m, 32); s2 += __shfl_xor(s2, m, 32); }
  float mean = s * (1.f / Cz);
  float var = s2 * (1.f / Cz) - mean * mean;
  float rstd = rsqrtf(var + EPSL);
#pragma unroll
  for (int i = 0; i < 4; ++i) {
    int c = lane * 4 + i;
    dst[c] = (h16)((vals[i] - mean) * rstd * g[c] + bt[c]);
  }
}

// ---------------- QKV GEMM: (Mwin x 128) @ (128 x 384), 16x64 per wave ------
__global__ void k_qkv(const h16* __restrict__ A, const h16* __restrict__ WtT,
                      const float* __restrict__ bias, h16* __restrict__ Qb,
                      h16* __restrict__ Kb, h16* __restrict__ Vt) {
  int wave = threadIdx.x >> 5, lane = threadIdx.x & 31;
  int m0 = blockIdx.x * 64 + wave * 16;
  int n0 = blockIdx.y * 64;
  v8f acc[4] = {};
  for (int kk = 0; kk < Cz; kk += 32) {
    v16h a = load_a(A, Cz, m0, kk);
#pragma unroll
    for (int t = 0; t < 4; ++t)
      acc[t] = wmma32(a, load_bT(WtT, Cz, kk, n0 + t * 16), acc[t]);
  }
  int colL = lane & 15, rbase = (lane >> 4) << 3;
#pragma unroll
  for (int t = 0; t < 4; ++t) {
    int col = n0 + t * 16 + colL;
    int mat = col >> 7, head = (col & 127) >> 5, hc = col & 31;
    float bb = bias[col];
    for (int r = 0; r < 8; ++r) {
      int row = m0 + rbase + r;
      int bn = row / Np, n = row % Np;
      float v = (n < Nw) ? (acc[t][r] + bb) : 0.f;
      if (mat == 0)
        Qb[(((size_t)bn * HEADS + head) * Np + n) * HD + hc] = (h16)(v * SCALE);
      else if (mat == 1)
        Kb[(((size_t)bn * HEADS + head) * Np + n) * HD + hc] = (h16)v;
      else  // V stored transposed (hd-major) so P@V B-fragments are contiguous
        Vt[(((size_t)bn * HEADS + head) * HD + hc) * Np + n] = (h16)v;
    }
  }
}

// ---------------- attention: per (window, head, 16-row tile) ----------------
__global__ void k_attn(const h16* __restrict__ Qb, const h16* __restrict__ Kb,
                       const h16* __restrict__ Vt, const float* __restrict__ rpb,
                       h16* __restrict__ AO) {
  __shared__ float S[16 * Np];
  __shared__ h16 P[16 * Np];
  int wave = threadIdx.x >> 5, lane = threadIdx.x & 31;
  constexpr int NT = Np / 16; // 26 col tiles
  int rt = blockIdx.x % NT;
  int head = (blockIdx.x / NT) % HEADS;
  int bn = blockIdx.x / (NT * HEADS);
  const h16* qb = Qb + ((size_t)bn * HEADS + head) * Np * HD;
  const h16* kb = Kb + ((size_t)bn * HEADS + head) * Np * HD;
  const h16* vtb = Vt + ((size_t)bn * HEADS + head) * HD * Np;
  int colL = lane & 15, rbase = (lane >> 4) << 3;

  // phase 1: S tile = q(16x32) @ k^T  (K rows are n-major -> load_bT)
  v16h aq = load_a(qb, HD, rt * 16, 0);
  for (int ct = wave; ct < NT; ct += 4) {
    v8f c = {};
    c = wmma32(aq, load_bT(kb, HD, 0, ct * 16), c);
#pragma unroll
    for (int r = 0; r < 8; ++r) S[(rbase + r) * Np + ct * 16 + colL] = c[r];
  }
  __syncthreads();

  // phase 2: + relative position bias + shift mask, pad -> -1e30
  int ih = (bn >> 3) & 7, iw = bn & 7;
  for (int e = threadIdx.x; e < 16 * Np; e += blockDim.x) {
    int r = e / Np, m = e % Np;
    int n = rt * 16 + r;
    if (m >= Nw || n >= Nw) { S[e] = -1e30f; continue; }
    // faithful quirky meshgrid decode: p -> (d, w', h') = (p/49,(p/7)%7,p%7)
    int pdn = n / 49, pwn = (n / 7) % 7, phn = n % 7;
    int pdm = m / 49, pwm = (m / 7) % 7, phm = m % 7;
    int idx = (pdn - pdm + 7) * 169 + (pwn - pwm + 6) * 13 + (phn - phm + 6);
    float add = rpb[idx * HEADS + head];
    // shift mask via region labels from the unshifted slice bounds (49/53)
    int ln = reg56(ih * 7 + pwn) * 3 + reg56(iw * 7 + phn);
    int lm = reg56(ih * 7 + pwm) * 3 + reg56(iw * 7 + phm);
    if (ln != lm) add -= 100.f;
    S[e] += add;
  }
  __syncthreads();

  // phase 3: row softmax, 8 lanes per row
  {
    int s8 = threadIdx.x & 7, r = threadIdx.x >> 3; // 16 rows x 8 lanes
    float mx = -3.0e38f;
    for (int m = s8; m < Np; m += 8) mx = fmaxf(mx, S[r * Np + m]);
    for (int mk = 1; mk < 8; mk <<= 1) mx = fmaxf(mx, __shfl_xor(mx, mk, 32));
    float sum = 0.f;
    for (int m = s8; m < Np; m += 8) {
      float e = expf(S[r * Np + m] - mx); S[r * Np + m] = e; sum += e;
    }
    for (int mk = 1; mk < 8; mk <<= 1) sum += __shfl_xor(sum, mk, 32);
    float inv = 1.f / sum;
    for (int m = s8; m < Np; m += 8) P[r * Np + m] = (h16)(S[r * Np + m] * inv);
  }
  __syncthreads();

  // phase 4: O(16x32) = P(16x416) @ V(416x32); 4 waves, K split 224/192
  {
    int khalf = wave >> 1;          // 0 -> K [0,224), 1 -> K [224,416)
    int n0 = (wave & 1) * 16;
    int kbeg = khalf ? 224 : 0, kend = khalf ? Np : 224;
    v8f acc = {};
    for (int kk = kbeg; kk < kend; kk += 32)
      acc = wmma32(load_a(P, Np, 0, kk), load_bT(vtb, Np, kk, n0), acc);
    if (khalf) {
#pragma unroll
      for (int r = 0; r < 8; ++r) S[(rbase + r) * 32 + n0 + colL] = acc[r];
    }
    __syncthreads();
    if (!khalf) {
#pragma unroll
      for (int r = 0; r < 8; ++r) {
        int n = rt * 16 + rbase + r;
        AO[((size_t)bn * Np + n) * Cz + head * HD + n0 + colL] =
            (h16)(acc[r] + S[(rbase + r) * 32 + n0 + colL]);
      }
    }
  }
}

// ---------------- proj GEMM + window reverse + un-roll + residual -----------
__global__ void k_proj(const h16* __restrict__ AO, const h16* __restrict__ WtT,
                       const float* __restrict__ bias, const float* __restrict__ xin,
                       float* __restrict__ x2) {
  int wave = threadIdx.x >> 5, lane = threadIdx.x & 31;
  int m0 = blockIdx.x * 64 + wave * 16;
  int n0 = blockIdx.y * 64;
  v8f acc[4] = {};
  for (int kk = 0; kk < Cz; kk += 32) {
    v16h a = load_a(AO, Cz, m0, kk);
#pragma unroll
    for (int t = 0; t < 4; ++t)
      acc[t] = wmma32(a, load_bT(WtT, Cz, kk, n0 + t * 16), acc[t]);
  }
  int colL = lane & 15, rbase = (lane >> 4) << 3;
#pragma unroll
  for (int t = 0; t < 4; ++t) {
    int col = n0 + t * 16 + colL;
    float bb = bias[col];
    for (int r = 0; r < 8; ++r) {
      int row = m0 + rbase + r;
      int bn = row / Np, n = row % Np;
      if (n >= Nw) continue;
      int b = bn >> 6, ih = (bn >> 3) & 7, iw = bn & 7;
      int td = n / 49, th = (n / 7) % 7, tw = n % 7;
      int d = td, hh = (ih * 7 + th + 3) % Hz, ww = (iw * 7 + tw + 3) % Wz;
      size_t gi = ((((size_t)b * Dz + d) * Hz + hh) * Wz + ww) * Cz + col;
      x2[gi] = xin[gi] + acc[t][r] + bb;
    }
  }
}

// ---------------- LN2 (identity layout) ----------------
__global__ void k_ln2(const float* __restrict__ x2, const float* __restrict__ g,
                      const float* __restrict__ bt, h16* __restrict__ out) {
  int wave = threadIdx.x >> 5, lane = threadIdx.x & 31;
  int t = blockIdx.x * (blockDim.x >> 5) + wave;
  if (t >= TOK) return;
  const float* src = x2 + (size_t)t * Cz;
  float vals[4], s = 0.f, s2 = 0.f;
#pragma unroll
  for (int i = 0; i < 4; ++i) {
    float v = src[lane * 4 + i]; vals[i] = v; s += v; s2 += v * v;
  }
  for (int m = 16; m >= 1; m >>= 1) { s += __shfl_xor(s, m, 32); s2 += __shfl_xor(s2, m, 32); }
  float mean = s * (1.f / Cz);
  float var = s2 * (1.f / Cz) - mean * mean;
  float rstd = rsqrtf(var + EPSL);
  h16* dst = out + (size_t)t * Cz;
#pragma unroll
  for (int i = 0; i < 4; ++i) {
    int c = lane * 4 + i;
    dst[c] = (h16)((vals[i] - mean) * rstd * g[c] + bt[c]);
  }
}

// ---------------- FC1 (+ exact GELU), 16x64 per wave ----------------
__global__ void k_fc1(const h16* __restrict__ A, const h16* __restrict__ WtT,
                      const float* __restrict__ bias, h16* __restrict__ Hb) {
  int wave = threadIdx.x >> 5, lane = threadIdx.x & 31;
  int m0 = blockIdx.x * 64 + wave * 16;
  int n0 = blockIdx.y * 64;
  v8f acc[4] = {};
  for (int kk = 0; kk < Cz; kk += 32) {
    v16h a = load_a(A, Cz, m0, kk);
#pragma unroll
    for (int t = 0; t < 4; ++t)
      acc[t] = wmma32(a, load_bT(WtT, Cz, kk, n0 + t * 16), acc[t]);
  }
  int colL = lane & 15, rbase = (lane >> 4) << 3;
#pragma unroll
  for (int t = 0; t < 4; ++t) {
    int col = n0 + t * 16 + colL;
    float bb = bias[col];
    for (int r = 0; r < 8; ++r) {
      float v = acc[t][r] + bb;
      float ge = 0.5f * v * (1.f + erff(v * 0.70710678118654752f));
      Hb[(size_t)(m0 + rbase + r) * HID + col] = (h16)ge;
    }
  }
}

// ---------------- FC2 + residual -> d_out, 16x64 per wave ----------------
__global__ void k_fc2(const h16* __restrict__ Hb, const h16* __restrict__ WtT,
                      const float* __restrict__ bias, const float* __restrict__ x2,
                      float* __restrict__ out) {
  int wave = threadIdx.x >> 5, lane = threadIdx.x & 31;
  int m0 = blockIdx.x * 64 + wave * 16;
  int n0 = blockIdx.y * 64;
  v8f acc[4] = {};
  for (int kk = 0; kk < HID; kk += 32) {
    v16h a = load_a(Hb, HID, m0, kk);
#pragma unroll
    for (int t = 0; t < 4; ++t)
      acc[t] = wmma32(a, load_bT(WtT, HID, kk, n0 + t * 16), acc[t]);
  }
  int colL = lane & 15, rbase = (lane >> 4) << 3;
#pragma unroll
  for (int t = 0; t < 4; ++t) {
    int col = n0 + t * 16 + colL;
    float bb = bias[col];
    for (int r = 0; r < 8; ++r) {
      size_t gi = (size_t)(m0 + rbase + r) * Cz + col;
      out[gi] = x2[gi] + acc[t][r] + bb;
    }
  }
}

extern "C" void kernel_launch(void* const* d_in, const int* in_sizes, int n_in,
                              void* d_out, int out_size, void* d_ws, size_t ws_size,
                              hipStream_t stream) {
  const float* x     = (const float*)d_in[0];
  const float* n1g   = (const float*)d_in[1];
  const float* n1b   = (const float*)d_in[2];
  const float* qkvw  = (const float*)d_in[3];
  const float* qkvb  = (const float*)d_in[4];
  const float* projw = (const float*)d_in[5];
  const float* projb = (const float*)d_in[6];
  const float* rpb   = (const float*)d_in[7];
  const float* n2g   = (const float*)d_in[8];
  const float* n2b   = (const float*)d_in[9];
  const float* fc1w  = (const float*)d_in[10];
  const float* fc1b  = (const float*)d_in[11];
  const float* fc2w  = (const float*)d_in[12];
  const float* fc2b  = (const float*)d_in[13];
  float* out = (float*)d_out;

  char* ws = (char*)d_ws;
  size_t off = 0;
  auto alloc = [&](size_t bytes) { size_t o = off; off += (bytes + 255) & ~(size_t)255; return o; };
  size_t oXW = alloc((size_t)Mwin * Cz * 2);           // windowed LN1 output (f16)
  size_t oQW = alloc((size_t)Cz * 384 * 2);            // transposed f16 weights
  size_t oPW = alloc((size_t)Cz * Cz * 2);
  size_t oF1 = alloc((size_t)Cz * HID * 2);
  size_t oF2 = alloc((size_t)HID * Cz * 2);
  size_t oQ  = alloc((size_t)NWIN * HEADS * Np * HD * 2);
  size_t oK  = alloc((size_t)NWIN * HEADS * Np * HD * 2);
  size_t oV  = alloc((size_t)NWIN * HEADS * Np * HD * 2); // transposed V
  size_t oX2 = alloc((size_t)TOK * Cz * 4);            // post-attn residual (f32)
  size_t oH  = alloc((size_t)TOK * HID * 2);           // MLP hidden (f16)
  size_t oAO  = oXW;  // reuse: xw dead after QKV GEMM
  size_t oXN2 = oQ;   // reuse: q dead after attention

  h16* XW  = (h16*)(ws + oXW);
  h16* QWt = (h16*)(ws + oQW);
  h16* PWt = (h16*)(ws + oPW);
  h16* F1t = (h16*)(ws + oF1);
  h16* F2t = (h16*)(ws + oF2);
  h16* Qb  = (h16*)(ws + oQ);
  h16* Kb  = (h16*)(ws + oK);
  h16* Vt  = (h16*)(ws + oV);
  float* X2 = (float*)(ws + oX2);
  h16* Hb  = (h16*)(ws + oH);
  h16* AO  = (h16*)(ws + oAO);
  h16* XN2 = (h16*)(ws + oXN2);

  // weight convert + transpose to N-major f16 (B fragments become contiguous)
  k_cvt_t<<<(Cz * 384 + 255) / 256, 256, 0, stream>>>(qkvw, QWt, Cz, 384);
  k_cvt_t<<<(Cz * Cz + 255) / 256, 256, 0, stream>>>(projw, PWt, Cz, Cz);
  k_cvt_t<<<(Cz * HID + 255) / 256, 256, 0, stream>>>(fc1w, F1t, Cz, HID);
  k_cvt_t<<<(HID * Cz + 255) / 256, 256, 0, stream>>>(fc2w, F2t, HID, Cz);

  // LN1 + roll + window partition (8 tokens per 256-thread block)
  k_ln1_window<<<Mwin / 8, 256, 0, stream>>>(x, n1g, n1b, XW);

  // QKV GEMM: 64x64 block tile (4 waves x 16x64)
  k_qkv<<<dim3(Mwin / 64, 384 / 64), 128, 0, stream>>>(XW, QWt, qkvb, Qb, Kb, Vt);

  // attention: 128 windows * 4 heads * 26 row tiles
  k_attn<<<NWIN * HEADS * (Np / 16), 128, 0, stream>>>(Qb, Kb, Vt, rpb, AO);

  // proj GEMM + reverse + residual
  k_proj<<<dim3(Mwin / 64, Cz / 64), 128, 0, stream>>>(AO, PWt, projb, x, X2);

  // LN2
  k_ln2<<<TOK / 8, 256, 0, stream>>>(X2, n2g, n2b, XN2);

  // MLP
  k_fc1<<<dim3(TOK / 64, HID / 64), 128, 0, stream>>>(XN2, F1t, fc1b, Hb);
  k_fc2<<<dim3(TOK / 64, Cz / 64), 128, 0, stream>>>(Hb, F2t, fc2b, X2, out);
}
